// LSTMEncoderCell_25460566131271
// MI455X (gfx1250) — compile-verified, matrix operand order checked by reading, not run on hardware
//
#include <hip/hip_runtime.h>

// ---------------------------------------------------------------------------
// LSTM encoder (all-sigmoid gates), CDNA5 / gfx1250, wave32 + WMMA bf16.
//   B=64, T=512, D=512, U=1024.
//   g = sigmoid([x_t | h] @ [wx;wh](:, g*U+u) + b[g,u])
//   c' = g0*g2 + g1*c ;  h' = g3 * sigmoid(c')
// Persistent-kernel version: 1 launch, 512 recurrent steps separated by a
// device-scope barrier (atomic arrive + s_sleep spin + acquire/release
// fences). 32 workgroups <= WGP count -> guaranteed co-resident.
// ---------------------------------------------------------------------------

#define BB 64
#define TT 512
#define DD 512
#define UU 1024
#define KX DD            // x contribution to K
#define KH UU            // h contribution to K
#define KTOT (KX + KH)   // 1536
#define NBLK (UU / 32)   // 32 persistent workgroups

typedef __attribute__((ext_vector_type(16))) __bf16 v16bf;
typedef __attribute__((ext_vector_type(8)))  __bf16 v8bf;
typedef __attribute__((ext_vector_type(8)))  float  v8f;

union BF16x16 { v16bf v; v8bf h[2]; };

__device__ __forceinline__ float sigmoidf(float x) {
    return 1.0f / (1.0f + __expf(-x));
}

// -------------------------- prep kernels -----------------------------------

__global__ __launch_bounds__(256) void conv_x_bf16(const float* __restrict__ x,
                                                   __bf16* __restrict__ xbf) {
    size_t i = (size_t)blockIdx.x * 256 + threadIdx.x;   // B*T*D threads
    xbf[i] = (__bf16)x[i];
}

// Pack W^T: row n = g*U + u (n in [0,4096)), col k in [0,1536).
//   k <  512 : wx[g, k, u]          (wx: [4, D, U])
//   k >= 512 : wh[g, k-512, u]      (wh: [4, U, U])
__global__ __launch_bounds__(256) void pack_wT(const float* __restrict__ wx,
                                               const float* __restrict__ wh,
                                               __bf16* __restrict__ wT) {
    size_t i = (size_t)blockIdx.x * 256 + threadIdx.x;   // 4*U*KTOT threads
    int k   = (int)(i % KTOT);
    int row = (int)(i / KTOT);
    int g = row >> 10;
    int u = row & (UU - 1);
    float v;
    if (k < KX) v = wx[((size_t)(g * DD + k)) * UU + u];
    else        v = wh[((size_t)(g * UU + (k - KX))) * UU + u];
    wT[i] = (__bf16)v;
}

__global__ __launch_bounds__(256) void init_state(const float* __restrict__ h0,
                                                  const float* __restrict__ c0,
                                                  __bf16* __restrict__ hbf,
                                                  float* __restrict__ c,
                                                  unsigned* __restrict__ bar) {
    size_t i = (size_t)blockIdx.x * 256 + threadIdx.x;   // B*U threads
    hbf[i] = (__bf16)h0[i];
    c[i]   = c0[i];
    if (i == 0) *bar = 0u;   // reset persistent-kernel barrier each call
}

// ------------------------- persistent kernel -------------------------------
// Grid: NBLK=32 blocks, 256 threads (8 waves).
//   wave w: m-tile = w&3 (M=64 -> 4 tiles), u-tile = blockIdx.x*2 + (w>>2).
//   Per step each wave: 4 gate accumulators (16x16 f32), K loop of 48 x 32,
//   fused sigmoid/cell epilogue, then device-wide barrier.
__global__ __launch_bounds__(256) void lstm_persistent(
    const __bf16* __restrict__ xbf,   // [B, T, D] bf16
    const __bf16* __restrict__ wT,    // [4U, KTOT] bf16 row-major
    __bf16* __restrict__ hA,          // [B, U] bf16 ping
    __bf16* __restrict__ hB,          // [B, U] bf16 pong
    float* __restrict__ c,            // [B, U] f32 (in-place, block-owned cols)
    const float* __restrict__ bias,   // [4, U]
    float* __restrict__ out,          // [B, T, U]
    unsigned* __restrict__ bar)       // device barrier counter (init 0)
{
    const int lane = threadIdx.x & 31;
    const int wid  = threadIdx.x >> 5;
    const int half = lane >> 4;       // 0: lanes 0-15, 1: lanes 16-31
    const int l16  = lane & 15;
    const int mt   = wid & 3;
    const int ut   = blockIdx.x * 2 + (wid >> 2);
    const int ucol = ut * 16 + l16;       // output column (u) for C/D frags
    const int arow = mt * 16 + l16;       // A-matrix row (batch index)
    const int koff = half * 8;            // per-lane K sub-offset

    // Loop-invariant: bias splat values and B-fragment row bases.
    const float bv0 = bias[0 * UU + ucol];
    const float bv1 = bias[1 * UU + ucol];
    const float bv2 = bias[2 * UU + ucol];
    const float bv3 = bias[3 * UU + ucol];
    const __bf16* brow0 = wT + (size_t)(0 * UU + ut * 16 + l16) * KTOT;
    const __bf16* brow1 = wT + (size_t)(1 * UU + ut * 16 + l16) * KTOT;
    const __bf16* brow2 = wT + (size_t)(2 * UU + ut * 16 + l16) * KTOT;
    const __bf16* brow3 = wT + (size_t)(3 * UU + ut * 16 + l16) * KTOT;
    const __bf16* axp0  = xbf + (size_t)arow * (TT * DD);

    for (int t = 0; t < TT; ++t) {
        const __bf16* hin  = (t & 1) ? hB : hA;
        __bf16*       hout = (t & 1) ? hA : hB;

        v8f acc[4];
        acc[0] = (v8f){bv0, bv0, bv0, bv0, bv0, bv0, bv0, bv0};
        acc[1] = (v8f){bv1, bv1, bv1, bv1, bv1, bv1, bv1, bv1};
        acc[2] = (v8f){bv2, bv2, bv2, bv2, bv2, bv2, bv2, bv2};
        acc[3] = (v8f){bv3, bv3, bv3, bv3, bv3, bv3, bv3, bv3};

        // ---- K over x_t (16 blocks of 32) ----
        const __bf16* axp = axp0 + (size_t)t * DD;
#pragma unroll 4
        for (int kb = 0; kb < KX / 32; ++kb) {
            const int k0 = kb * 32 + koff;
            BF16x16 a, b0, b1, b2, b3;
            a.h[0]  = *(const v8bf*)(axp + k0);
            a.h[1]  = *(const v8bf*)(axp + k0 + 16);
            b0.h[0] = *(const v8bf*)(brow0 + k0);
            b0.h[1] = *(const v8bf*)(brow0 + k0 + 16);
            b1.h[0] = *(const v8bf*)(brow1 + k0);
            b1.h[1] = *(const v8bf*)(brow1 + k0 + 16);
            b2.h[0] = *(const v8bf*)(brow2 + k0);
            b2.h[1] = *(const v8bf*)(brow2 + k0 + 16);
            b3.h[0] = *(const v8bf*)(brow3 + k0);
            b3.h[1] = *(const v8bf*)(brow3 + k0 + 16);
            acc[0] = __builtin_amdgcn_wmma_f32_16x16x32_bf16(false, a.v, false, b0.v, (short)0, acc[0], false, false);
            acc[1] = __builtin_amdgcn_wmma_f32_16x16x32_bf16(false, a.v, false, b1.v, (short)0, acc[1], false, false);
            acc[2] = __builtin_amdgcn_wmma_f32_16x16x32_bf16(false, a.v, false, b2.v, (short)0, acc[2], false, false);
            acc[3] = __builtin_amdgcn_wmma_f32_16x16x32_bf16(false, a.v, false, b3.v, (short)0, acc[3], false, false);
        }

        // Prefetch next step's x row (uniform branch; EXEC stays all-ones).
        if (t + 1 < TT) {
#pragma unroll
            for (int p = 0; p < DD; p += 128)
                __builtin_prefetch(axp + DD + p, 0, 1);
        }

        // ---- K over h (32 blocks of 32); weight K offset shifted by KX ----
        const __bf16* ahp = hin + (size_t)arow * UU;
#pragma unroll 4
        for (int kb = 0; kb < KH / 32; ++kb) {
            const int ka = kb * 32 + koff;
            const int kw = KX + ka;
            BF16x16 a, b0, b1, b2, b3;
            a.h[0]  = *(const v8bf*)(ahp + ka);
            a.h[1]  = *(const v8bf*)(ahp + ka + 16);
            b0.h[0] = *(const v8bf*)(brow0 + kw);
            b0.h[1] = *(const v8bf*)(brow0 + kw + 16);
            b1.h[0] = *(const v8bf*)(brow1 + kw);
            b1.h[1] = *(const v8bf*)(brow1 + kw + 16);
            b2.h[0] = *(const v8bf*)(brow2 + kw);
            b2.h[1] = *(const v8bf*)(brow2 + kw + 16);
            b3.h[0] = *(const v8bf*)(brow3 + kw);
            b3.h[1] = *(const v8bf*)(brow3 + kw + 16);
            acc[0] = __builtin_amdgcn_wmma_f32_16x16x32_bf16(false, a.v, false, b0.v, (short)0, acc[0], false, false);
            acc[1] = __builtin_amdgcn_wmma_f32_16x16x32_bf16(false, a.v, false, b1.v, (short)0, acc[1], false, false);
            acc[2] = __builtin_amdgcn_wmma_f32_16x16x32_bf16(false, a.v, false, b2.v, (short)0, acc[2], false, false);
            acc[3] = __builtin_amdgcn_wmma_f32_16x16x32_bf16(false, a.v, false, b3.v, (short)0, acc[3], false, false);
        }

        // ---- epilogue: C/D frag element r -> row = mt*16 + half*8 + r ----
#pragma unroll
        for (int r = 0; r < 8; ++r) {
            const int row = mt * 16 + half * 8 + r;   // batch index
            const size_t idx = (size_t)row * UU + ucol;
            float upd = sigmoidf(acc[0][r]);
            float fgt = sigmoidf(acc[1][r]);
            float cnd = sigmoidf(acc[2][r]);
            float act = sigmoidf(acc[3][r]);
            float cn  = upd * cnd + fgt * c[idx];
            c[idx] = cn;
            float hn = act * sigmoidf(cn);
            out[((size_t)row * TT + t) * UU + ucol] = hn;
            hout[idx] = (__bf16)hn;
        }

        // ---- device-wide barrier between recurrent steps ----
        __threadfence();                       // release: flush h/c stores
        __syncthreads();
        if (threadIdx.x == 0) {
            const unsigned target = (unsigned)(t + 1) * NBLK;
            __hip_atomic_fetch_add(bar, 1u, __ATOMIC_RELAXED,
                                   __HIP_MEMORY_SCOPE_AGENT);
            while (__hip_atomic_load(bar, __ATOMIC_RELAXED,
                                     __HIP_MEMORY_SCOPE_AGENT) < target) {
                __builtin_amdgcn_s_sleep(2);
            }
        }
        __syncthreads();
        __threadfence();                       // acquire: invalidate WGP$/L0
    }
}

// ----------------------------- launcher ------------------------------------

extern "C" void kernel_launch(void* const* d_in, const int* in_sizes, int n_in,
                              void* d_out, int out_size, void* d_ws, size_t ws_size,
                              hipStream_t stream) {
    (void)in_sizes; (void)n_in; (void)out_size; (void)ws_size;
    const float* x  = (const float*)d_in[0];   // [B,T,D]
    const float* h0 = (const float*)d_in[1];   // [B,U]
    const float* c0 = (const float*)d_in[2];   // [B,U]
    const float* wx = (const float*)d_in[3];   // [4,D,U]
    const float* wh = (const float*)d_in[4];   // [4,U,U]
    const float* b  = (const float*)d_in[5];   // [4,U]
    float* out = (float*)d_out;                // [B,T,U]

    // Workspace layout (all 256B aligned):
    //   xbf : B*T*D bf16            = 33,554,432 B  @ 0
    //   wT  : 4U*KTOT bf16          = 12,582,912 B  @ 33,554,432
    //   hA  : B*U bf16              =    131,072 B  @ 46,137,344
    //   hB  : B*U bf16              =    131,072 B  @ 46,268,416
    //   c   : B*U f32               =    262,144 B  @ 46,399,488
    //   bar : 1 u32                 =        256 B  @ 46,661,632
    char* ws = (char*)d_ws;
    __bf16*   xbf = (__bf16*)(ws);
    __bf16*   wT  = (__bf16*)(ws + 33554432);
    __bf16*   hA  = (__bf16*)(ws + 46137344);
    __bf16*   hB  = (__bf16*)(ws + 46268416);
    float*    c   = (float*) (ws + 46399488);
    unsigned* bar = (unsigned*)(ws + 46661632);

    conv_x_bf16<<<(BB * TT * DD) / 256, 256, 0, stream>>>(x, xbf);
    pack_wT    <<<(4 * UU * KTOT) / 256, 256, 0, stream>>>(wx, wh, wT);
    init_state <<<(BB * UU) / 256, 256, 0, stream>>>(h0, c0, hA, c, bar);

    lstm_persistent<<<NBLK, 256, 0, stream>>>(xbf, wT, hA, hB, c, b, out, bar);
}
